// NeuralSymbolicLayer_60636348285697
// MI455X (gfx1250) — matrix operand order repaired and checked by reading, not compiled, not run on hardware
//
#include <hip/hip_runtime.h>

typedef __attribute__((ext_vector_type(16))) __bf16 v16bf;
typedef __attribute__((ext_vector_type(8)))  __bf16 v8bf;
typedef __attribute__((ext_vector_type(8)))  float  v8f;
typedef __attribute__((ext_vector_type(8)))  unsigned v8u;

#define D_    2048
#define NP_   64
#define NC_   32
#define ITERS_ 3
#define NTOK_ 16384
#define WAVES_ 8
#define TPW_  16      // tokens per wave
#define TPB_  128     // tokens per block
#define NBLK_ (NTOK_/TPB_)
#define PWS_  2056    // padded row stride (bf16) for staged prop_w: bank-conflict-free

// union region layout (bytes)
#define U_BYTES   263168            // 64 * PWS_ * 2  (staged prop_w bf16)
#define U_OFF_W1  0                 // 128*128 bf16 = 32768
#define U_OFF_W2  32768             // 64*128 bf16  = 16384
#define U_OFF_PR  49152             // props f32: 8*16*64*4 = 32768
#define U_OFF_CK  81920             // chk  f32: 8*16*32*4 = 16384
#define U_OFF_VI  98304             // viol bf16: 8*16*32*2 = 8192

static constexpr size_t OFF_P = (size_t)NTOK_ * D_;
static constexpr size_t OFF_S = OFF_P + (size_t)NTOK_ * NP_;
static constexpr size_t OFF_C = OFF_S + (size_t)NTOK_ * NC_;
static constexpr size_t OFF_L = OFF_C + (size_t)NTOK_;

// f32 -> bf16, round-to-nearest, 2 VALU
__device__ __forceinline__ __bf16 f2bf(float f) {
  unsigned u = __builtin_bit_cast(unsigned, f) + 0x8000u;
  return __builtin_bit_cast(__bf16, (unsigned short)(u >> 16));
}
// two f32 -> packed bf16 pair: 2x v_add + 1x v_perm_b32
__device__ __forceinline__ unsigned f2bf2u(float a, float b) {
  unsigned ua = __builtin_bit_cast(unsigned, a) + 0x8000u;
  unsigned ub = __builtin_bit_cast(unsigned, b) + 0x8000u;
  return __builtin_amdgcn_perm(ub, ua, 0x07060302u);
}
__device__ __forceinline__ v16bf pack16(float4 q0, float4 q1, float4 q2, float4 q3) {
  v8u u;
  u[0] = f2bf2u(q0.x, q0.y); u[1] = f2bf2u(q0.z, q0.w);
  u[2] = f2bf2u(q1.x, q1.y); u[3] = f2bf2u(q1.z, q1.w);
  u[4] = f2bf2u(q2.x, q2.y); u[5] = f2bf2u(q2.z, q2.w);
  u[6] = f2bf2u(q3.x, q3.y); u[7] = f2bf2u(q3.z, q3.w);
  return __builtin_bit_cast(v16bf, u);
}
__device__ __forceinline__ float sigm(float x) { return 1.0f / (1.0f + __expf(-x)); }
__device__ __forceinline__ float gelu_fast(float x) {
  float t = x * (1.5957691216057308f + 0.07135481627f * x * x);
  return x * sigm(t);
}
__device__ __forceinline__ v8f wmma_bf16(v16bf a, v16bf b, v8f c) {
  return __builtin_amdgcn_wmma_f32_16x16x32_bf16(false, a, false, b, (short)0, c, false, false);
}

__device__ __forceinline__ v16bf load_a16_global(const float* rowk, int lane) {
  const int koff = (lane < 16) ? 0 : 8;
  const float4* p0 = (const float4*)(rowk + koff);
  const float4* p1 = (const float4*)(rowk + 16 + koff);
  return pack16(p0[0], p0[1], p1[0], p1[1]);
}
__device__ __forceinline__ v16bf load_b16_global(const float* w, int ld, int n0, int kk, int lane) {
  const int n  = n0 + (lane & 15);
  const int kb = kk + ((lane < 16) ? 0 : 8);
  const float4* p = (const float4*)(w + (size_t)n * ld + kb);
  const float4* p1 = (const float4*)(w + (size_t)n * ld + kb + 16);
  return pack16(p[0], p[1], p1[0], p1[1]);
}
__device__ __forceinline__ v16bf load_a16_lds(const __bf16* h, int ldh, int kk, int lane) {
  const int m = lane & 15;
  const int koff = (lane < 16) ? 0 : 8;
  v8bf lo = *(const v8bf*)(h + m * ldh + kk + koff);
  v8bf hi = *(const v8bf*)(h + m * ldh + kk + 16 + koff);
  return __builtin_shufflevector(lo, hi, 0, 1, 2, 3, 4, 5, 6, 7,
                                 8, 9, 10, 11, 12, 13, 14, 15);
}
__device__ __forceinline__ v16bf load_b16_lds(const __bf16* w, int ldw, int n0, int kk, int lane) {
  const int n  = n0 + (lane & 15);
  const int kb = kk + ((lane < 16) ? 0 : 8);
  v8bf lo = *(const v8bf*)(w + n * ldw + kb);
  v8bf hi = *(const v8bf*)(w + n * ldw + kb + 16);
  return __builtin_shufflevector(lo, hi, 0, 1, 2, 3, 4, 5, 6, 7,
                                 8, 9, 10, 11, 12, 13, 14, 15);
}

__global__ __launch_bounds__(256, 2) void nsl_fused_kernel(
    const float* __restrict__ x, const float* __restrict__ prop_w,
    const float* __restrict__ prop_b, const float* __restrict__ constraint_mat,
    const float* __restrict__ polarity, const float* __restrict__ corr_w1,
    const float* __restrict__ corr_b1, const float* __restrict__ corr_w2,
    const float* __restrict__ corr_b2, const float* __restrict__ cons_w,
    const float* __restrict__ cons_b, const float* __restrict__ emb_w,
    const float* __restrict__ emb_b, const float* __restrict__ norm_w,
    float* __restrict__ out, float* __restrict__ ws) {
  // phase-aliased union region: staged prop_w bf16 (phase 1) OR iteration arrays
  __shared__ __align__(16) char uShared[U_BYTES];
  __shared__ __align__(16) __bf16 sHB[WAVES_][TPW_ * 128];   // h / props bf16 (A src)
  __shared__ __align__(16) __bf16 sWPc[NC_ * NP_];           // (W-P)[c][j] bf16
  __shared__ __align__(16) __bf16 sWtB[NP_ * NC_];           // W^T[j][c] bf16
  __shared__ __align__(16) float  sPsum[NC_];
  __shared__ __align__(16) float  sRed[WAVES_][32 * 8];
  __shared__ float sCons[WAVES_][TPW_];
  __shared__ float sRms[WAVES_][TPW_];
  __shared__ float sLossW[WAVES_];

  __bf16* pwLDS = (__bf16*)uShared;                          // [64][PWS_]
  __bf16* sW1   = (__bf16*)(uShared + U_OFF_W1);             // [128][128]
  __bf16* sW2   = (__bf16*)(uShared + U_OFF_W2);             // [64][128]
  float*  sPropsAll = (float*)(uShared + U_OFF_PR);          // [8][16*64]
  float*  sChkAll   = (float*)(uShared + U_OFF_CK);          // [8][16*32]
  __bf16* sViolAll  = (__bf16*)(uShared + U_OFF_VI);         // [8][16*32]

  const int tid  = threadIdx.x;
  const int wave = tid >> 5;
  const int lane = tid & 31;
  const int mBase = (lane >> 4) << 3;     // token offset for C-layout rows
  const int t0 = blockIdx.x * TPB_ + wave * TPW_;
  const float invNP = 1.0f / (float)NP_;
  const float invNC = 1.0f / ((float)NC_ + 1e-6f);

  float*  myProps = sPropsAll + wave * (TPW_ * NP_);
  float*  myChk   = sChkAll + wave * (TPW_ * NC_);
  __bf16* myViol  = sViolAll + wave * (TPW_ * NC_);
  __bf16* myHB    = &sHB[wave][0];

  // ---- stage constraint weights + prop_w (bf16, padded rows) into LDS ----
  for (int i = tid; i < NC_ * NP_; i += 256) {
    int c = i >> 6, j = i & 63;
    float wv = sigm(constraint_mat[i]);
    float pv = sigm(polarity[i]);
    sWPc[i] = f2bf(wv - pv);
    sWtB[j * NC_ + c] = f2bf(wv);
  }
  if (tid < NC_) {
    float s = 0.f;
    for (int j = 0; j < NP_; j++) s += sigm(polarity[tid * NP_ + j]);
    sPsum[tid] = s;
  }
  {
    const float2* pwv = (const float2*)prop_w;
    for (int i = tid; i < NP_ * D_ / 2; i += 256) {
      int n = i >> 10, k2 = i & 1023;                // 1024 pairs per row
      float2 q = pwv[i];
      ((unsigned*)(pwLDS + n * PWS_))[k2] = f2bf2u(q.x, q.y);
    }
  }
  __syncthreads();

  // ---- phase 1 (software pipelined): props = sigmoid(x @ prop_w^T + b) ----
  v8f acc1[4];
  {
    v8f z = {};
    acc1[0] = z; acc1[1] = z; acc1[2] = z; acc1[3] = z;
    const float* xrow = x + (size_t)(t0 + (lane & 15)) * D_;
    v16bf a_cur = load_a16_global(xrow, lane);
    for (int kk = 0; kk < D_ - 32; kk += 32) {
      __builtin_prefetch(xrow + kk + 288, 0, 1);
      v16bf b0 = load_b16_lds(pwLDS, PWS_, 0,  kk, lane);
      v16bf b1 = load_b16_lds(pwLDS, PWS_, 16, kk, lane);
      v16bf b2 = load_b16_lds(pwLDS, PWS_, 32, kk, lane);
      v16bf b3 = load_b16_lds(pwLDS, PWS_, 48, kk, lane);
      v16bf a_next = load_a16_global(xrow + kk + 32, lane);   // overlaps with WMMAs
      acc1[0] = wmma_bf16(a_cur, b0, acc1[0]);
      acc1[1] = wmma_bf16(a_cur, b1, acc1[1]);
      acc1[2] = wmma_bf16(a_cur, b2, acc1[2]);
      acc1[3] = wmma_bf16(a_cur, b3, acc1[3]);
      a_cur = a_next;
    }
    { // peeled last K-chunk
      const int kk = D_ - 32;
      v16bf b0 = load_b16_lds(pwLDS, PWS_, 0,  kk, lane);
      v16bf b1 = load_b16_lds(pwLDS, PWS_, 16, kk, lane);
      v16bf b2 = load_b16_lds(pwLDS, PWS_, 32, kk, lane);
      v16bf b3 = load_b16_lds(pwLDS, PWS_, 48, kk, lane);
      acc1[0] = wmma_bf16(a_cur, b0, acc1[0]);
      acc1[1] = wmma_bf16(a_cur, b1, acc1[1]);
      acc1[2] = wmma_bf16(a_cur, b2, acc1[2]);
      acc1[3] = wmma_bf16(a_cur, b3, acc1[3]);
    }
  }
  __syncthreads();   // all waves done reading pwLDS; union region is reusable

  // ---- epilogue of phase 1 + stage corr weights (alias pwLDS region) ----
#pragma unroll
  for (int nt = 0; nt < 4; nt++) {
    int col = nt * 16 + (lane & 15);
    float bias = prop_b[col];
#pragma unroll
    for (int r = 0; r < 8; r++) {
      float p = sigm(acc1[nt][r] + bias);
      myProps[(r + mBase) * NP_ + col] = p;
      myHB[(r + mBase) * 128 + col] = f2bf(p);
    }
  }
  {
    const float2* w1v = (const float2*)corr_w1;
    unsigned* s1u = (unsigned*)sW1;
    for (int i = tid; i < 128 * 128 / 2; i += 256) {
      float2 q = w1v[i];
      s1u[i] = f2bf2u(q.x, q.y);
    }
    const float2* w2v = (const float2*)corr_w2;
    unsigned* s2u = (unsigned*)sW2;
    for (int i = tid; i < 64 * 128 / 2; i += 256) {
      float2 q = w2v[i];
      s2u[i] = f2bf2u(q.x, q.y);
    }
  }
  __syncthreads();

  // ---- iterative correction (all matrix work on WMMA) ----
  for (int it = 0; it < ITERS_; it++) {
    // check GEMM: sat*NP = props @ (W-P)^T + Psum ; violation = 1 - sat
    {
      v8f accc[2];
#pragma unroll
      for (int nt = 0; nt < 2; nt++) {
        float ps = sPsum[nt * 16 + (lane & 15)];
#pragma unroll
        for (int r = 0; r < 8; r++) accc[nt][r] = ps;
      }
      v16bf a0 = load_a16_lds(myHB, 128, 0, lane);
      v16bf a1 = load_a16_lds(myHB, 128, 32, lane);
      v16bf b00 = load_b16_lds(sWPc, NP_, 0,  0, lane);
      v16bf b01 = load_b16_lds(sWPc, NP_, 16, 0, lane);
      v16bf b10 = load_b16_lds(sWPc, NP_, 0,  32, lane);
      v16bf b11 = load_b16_lds(sWPc, NP_, 16, 32, lane);
      accc[0] = wmma_bf16(a0, b00, accc[0]);
      accc[1] = wmma_bf16(a0, b01, accc[1]);
      accc[0] = wmma_bf16(a1, b10, accc[0]);
      accc[1] = wmma_bf16(a1, b11, accc[1]);
#pragma unroll
      for (int nt = 0; nt < 2; nt++) {
        int col = nt * 16 + (lane & 15);
#pragma unroll
        for (int r = 0; r < 8; r++)
          myViol[(r + mBase) * NC_ + col] = f2bf(1.0f - accc[nt][r] * invNP);
      }
    }
    __syncthreads();
    // viol GEMM: h2 = (violation @ W) / (NC+eps)
    {
      v8f z = {};
      v8f accv[4] = {z, z, z, z};
      v16bf a = load_a16_lds(myViol, NC_, 0, lane);
      v16bf bb[4];
#pragma unroll
      for (int nt = 0; nt < 4; nt++) bb[nt] = load_b16_lds(sWtB, NC_, nt * 16, 0, lane);
#pragma unroll
      for (int nt = 0; nt < 4; nt++) accv[nt] = wmma_bf16(a, bb[nt], accv[nt]);
#pragma unroll
      for (int nt = 0; nt < 4; nt++) {
        int col = nt * 16 + (lane & 15);
#pragma unroll
        for (int r = 0; r < 8; r++)
          myHB[(r + mBase) * 128 + 64 + col] = f2bf(accv[nt][r] * invNC);
      }
    }
    __syncthreads();
    // corr1: [16,128] = h @ W1^T, gelu
    {
      v8f z = {};
      v8f a1[8] = {z, z, z, z, z, z, z, z};
#pragma unroll
      for (int kk = 0; kk < 128; kk += 32) {
        v16bf a = load_a16_lds(myHB, 128, kk, lane);
        v16bf bb[8];
#pragma unroll
        for (int nt = 0; nt < 8; nt++) bb[nt] = load_b16_lds(sW1, 128, nt * 16, kk, lane);
#pragma unroll
        for (int nt = 0; nt < 8; nt++) a1[nt] = wmma_bf16(a, bb[nt], a1[nt]);
      }
#pragma unroll
      for (int nt = 0; nt < 8; nt++) {
        int col = nt * 16 + (lane & 15);
        float b1 = corr_b1[col];
#pragma unroll
        for (int r = 0; r < 8; r++)
          myHB[(r + mBase) * 128 + col] = f2bf(gelu_fast(a1[nt][r] + b1));
      }
    }
    __syncthreads();
    // corr2: [16,64] = g @ W2^T -> sigmoid -> new props (f32 + bf16)
    {
      v8f z = {};
      v8f a2[4] = {z, z, z, z};
#pragma unroll
      for (int kk = 0; kk < 128; kk += 32) {
        v16bf a = load_a16_lds(myHB, 128, kk, lane);
        v16bf bb[4];
#pragma unroll
        for (int nt = 0; nt < 4; nt++) bb[nt] = load_b16_lds(sW2, 128, nt * 16, kk, lane);
#pragma unroll
        for (int nt = 0; nt < 4; nt++) a2[nt] = wmma_bf16(a, bb[nt], a2[nt]);
      }
#pragma unroll
      for (int nt = 0; nt < 4; nt++) {
        int col = nt * 16 + (lane & 15);
        float b2 = corr_b2[col];
#pragma unroll
        for (int r = 0; r < 8; r++) {
          float p = sigm(a2[nt][r] + b2);
          myProps[(r + mBase) * NP_ + col] = p;
          myHB[(r + mBase) * 128 + col] = f2bf(p);
        }
      }
    }
    __syncthreads();
  }

  // ---- final satisfaction via WMMA check ----
  {
    v8f accc[2];
#pragma unroll
    for (int nt = 0; nt < 2; nt++) {
      float ps = sPsum[nt * 16 + (lane & 15)];
#pragma unroll
      for (int r = 0; r < 8; r++) accc[nt][r] = ps;
    }
    v16bf a0 = load_a16_lds(myHB, 128, 0, lane);
    v16bf a1 = load_a16_lds(myHB, 128, 32, lane);
    v16bf b00 = load_b16_lds(sWPc, NP_, 0,  0, lane);
    v16bf b01 = load_b16_lds(sWPc, NP_, 16, 0, lane);
    v16bf b10 = load_b16_lds(sWPc, NP_, 0,  32, lane);
    v16bf b11 = load_b16_lds(sWPc, NP_, 16, 32, lane);
    accc[0] = wmma_bf16(a0, b00, accc[0]);
    accc[1] = wmma_bf16(a0, b01, accc[1]);
    accc[0] = wmma_bf16(a1, b10, accc[0]);
    accc[1] = wmma_bf16(a1, b11, accc[1]);
#pragma unroll
    for (int nt = 0; nt < 2; nt++) {
      int col = nt * 16 + (lane & 15);
#pragma unroll
      for (int r = 0; r < 8; r++)
        myChk[(r + mBase) * NC_ + col] = accc[nt][r] * invNP;
    }
  }
  __syncthreads();
  { // outputs: satisfaction, consistency, loss partial, props
    float* satOut = out + OFF_S + (size_t)t0 * NC_;
    for (int i = lane; i < TPW_ * NC_; i += 32) satOut[i] = myChk[i];
    if (lane < 16) {
      int t = lane;
      float acc = 0.f, vsum = 0.f;
      for (int c = 0; c < NC_; c++) {
        float s = myChk[t * NC_ + c];
        acc += s * cons_w[c];
        vsum += (1.0f - s);
      }
      float cons = sigm(acc + cons_b[0]);
      sCons[wave][t] = cons;
      out[OFF_C + t0 + t] = cons;
      sRed[wave][lane] = vsum;
    }
  }
  __syncthreads();
  if (lane == 0) {
    float s = 0.f;
    for (int t = 0; t < TPW_; t++) s += sRed[wave][t];
    sLossW[wave] = s;
  }
  {
    const float4* ps = (const float4*)myProps;
    float4* po = (float4*)(out + OFF_P + (size_t)t0 * NP_);
    for (int i = lane; i < TPW_ * NP_ / 4; i += 32) po[i] = ps[i];
  }
  __syncthreads();
  if (tid == 0) {
    float s = 0.f;
    for (int w = 0; w < WAVES_; w++) s += sLossW[w];
    ws[blockIdx.x] = s;
  }

  // ---- symbolic = props @ emb_w^T ; y = x + symbolic*consistency (pipelined B) ----
  {
    v16bf aA0 = load_a16_lds(myHB, 128, 0, lane);   // props bf16 cols 0..63
    v16bf aA1 = load_a16_lds(myHB, 128, 32, lane);
    float cr[8];
#pragma unroll
    for (int r = 0; r < 8; r++) cr[r] = sCons[wave][r + mBase];
    float ss[8] = {0.f, 0.f, 0.f, 0.f, 0.f, 0.f, 0.f, 0.f};
    v16bf b0 = load_b16_global(emb_w, NP_, 0, 0, lane);
    v16bf b1 = load_b16_global(emb_w, NP_, 0, 32, lane);
    for (int n0 = 0; n0 < D_; n0 += 16) {
      v16bf nb0 = b0, nb1 = b1;
      if (n0 + 16 < D_) {                      // prefetch next tile's B fragments
        nb0 = load_b16_global(emb_w, NP_, n0 + 16, 0, lane);
        nb1 = load_b16_global(emb_w, NP_, n0 + 16, 32, lane);
      }
      v8f acc = {};
      acc = wmma_bf16(aA0, b0, acc);
      acc = wmma_bf16(aA1, b1, acc);
      int col = n0 + (lane & 15);
      float eb = emb_b[col];
#pragma unroll
      for (int r = 0; r < 8; r++) {
        size_t idx = (size_t)(t0 + r + mBase) * D_ + col;
        float yv = x[idx] + (acc[r] + eb) * cr[r];
        out[idx] = yv;
        ss[r] += yv * yv;
      }
      b0 = nb0; b1 = nb1;
    }
#pragma unroll
    for (int r = 0; r < 8; r++) sRed[wave][lane * 8 + r] = ss[r];
  }
  __syncthreads();
  if (lane < 16) {
    int t = lane;
    int lbase = (t < 8) ? 0 : 16;
    int rr = t & 7;
    float s = 0.f;
    for (int l = 0; l < 16; l++) s += sRed[wave][(lbase + l) * 8 + rr];
    sRms[wave][t] = rsqrtf(s * (1.0f / D_) + 1e-6f);
  }
  __syncthreads();

  // ---- normalize pass: y *= rms * norm_w (coalesced re-sweep) ----
  for (int t = 0; t < TPW_; t++) {
    float scale = sRms[wave][t];
    float4* yrow = (float4*)(out + (size_t)(t0 + t) * D_);
    const float4* nw = (const float4*)norm_w;
    for (int i = lane; i < D_ / 4; i += 32) {
      float4 v = yrow[i];
      float4 w4 = nw[i];
      v.x *= scale * w4.x; v.y *= scale * w4.y;
      v.z *= scale * w4.z; v.w *= scale * w4.w;
      yrow[i] = v;
    }
  }
}

__global__ void nsl_finalize_loss(const float* __restrict__ ws, float* __restrict__ out) {
  __shared__ float sm[NBLK_];
  int tid = threadIdx.x;
  sm[tid] = ws[tid];
  __syncthreads();
  for (int s = NBLK_ / 2; s > 0; s >>= 1) {
    if (tid < s) sm[tid] += sm[tid + s];
    __syncthreads();
  }
  if (tid == 0) out[OFF_L] = sm[0] * (1.0f / ((float)NTOK_ * (float)NC_));
}

extern "C" void kernel_launch(void* const* d_in, const int* in_sizes, int n_in,
                              void* d_out, int out_size, void* d_ws, size_t ws_size,
                              hipStream_t stream) {
  (void)in_sizes; (void)n_in; (void)out_size; (void)ws_size;
  const float* x              = (const float*)d_in[0];
  const float* prop_w         = (const float*)d_in[1];
  const float* prop_b         = (const float*)d_in[2];
  const float* constraint_mat = (const float*)d_in[3];
  const float* polarity       = (const float*)d_in[4];
  const float* corr_w1        = (const float*)d_in[5];
  const float* corr_b1        = (const float*)d_in[6];
  const float* corr_w2        = (const float*)d_in[7];
  const float* corr_b2        = (const float*)d_in[8];
  const float* cons_w         = (const float*)d_in[9];
  const float* cons_b         = (const float*)d_in[10];
  const float* emb_w          = (const float*)d_in[11];
  const float* emb_b          = (const float*)d_in[12];
  const float* norm_w         = (const float*)d_in[13];
  float* out = (float*)d_out;
  float* ws  = (float*)d_ws;

  nsl_fused_kernel<<<NBLK_, 256, 0, stream>>>(
      x, prop_w, prop_b, constraint_mat, polarity, corr_w1, corr_b1,
      corr_w2, corr_b2, cons_w, cons_b, emb_w, emb_b, norm_w, out, ws);
  nsl_finalize_loss<<<1, NBLK_, 0, stream>>>(ws, out);
}
